// FullGap_62955630625135
// MI455X (gfx1250) — matrix operand order repaired and checked by reading, not compiled, not run on hardware
//
#include <hip/hip_runtime.h>

#define N_ATOMS   32768
#define N_SUPPORT 8192
#define D_FEAT    256
#define N_STRUCT  512
#define N_TRAIN   256

// main-kernel tiling
#define STAGE_COLS 64                      // support rows staged per TDM copy
#define ROW_BYTES  512                     // 256 bf16 per support row
#define PAD_BYTES  16                      // TDM pad -> bank-conflict-free ds reads
#define LDS_ROW    (ROW_BYTES + PAD_BYTES) // 528
#define BUF_BYTES  (STAGE_COLS * LDS_ROW)  // 33792
#define N_STAGES   (N_SUPPORT / STAGE_COLS)

typedef __attribute__((ext_vector_type(16))) __bf16        v16bf;
typedef __attribute__((ext_vector_type(8)))  float         v8f;
typedef __attribute__((ext_vector_type(4))) unsigned int   u32x4;
typedef __attribute__((ext_vector_type(8)))  int           i32x8;
typedef __attribute__((ext_vector_type(4)))  int           i32x4;

union Frag {
    v16bf v;
    u32x4 u[2];
};

// float -> bf16 round-to-nearest-even, raw bits
static __device__ __forceinline__ unsigned short f2bf(float f) {
    unsigned int u = __float_as_uint(f);
    u += 0x7FFFu + ((u >> 16) & 1u);
    return (unsigned short)(u >> 16);
}

// ---------------------------------------------------------------------------
// TDM: copy STAGE_COLS x 512B tile (row stride 512B in global) into LDS with
// 16B padding appended per row (LDS row stride 528B). data_size = 8 bytes.
// D# group0: count=1, lds_addr, global_addr, type=2.
// D# group1: data_size=3(8B) | pad_enable | pad_interval=6(512B) | pad_amount=3(16B);
//            tensor_dim0=64, tensor_dim1=64, tile_dim0=64, tile_dim1=64,
//            tensor_dim0_stride=64 (units of 8B = 512B).
// ---------------------------------------------------------------------------
static __device__ __forceinline__ void tdm_load_tile(unsigned lds_addr,
                                                     const unsigned short* gsrc) {
    const unsigned long long ga = (unsigned long long)(uintptr_t)gsrc;

    u32x4 g0;
    g0[0] = 0x1u;                                            // count=1, user mode
    g0[1] = lds_addr;                                        // LDS byte address
    g0[2] = (unsigned)(ga & 0xFFFFFFFFu);                    // global_addr[31:0]
    g0[3] = (unsigned)((ga >> 32) & 0x01FFFFFFu) | 0x80000000u; // [56:32] | type=2

    i32x8 g1;
    g1[0] = (int)((3u << 16) | (1u << 20) | (6u << 22) | (3u << 25)); // 0x07930000
    g1[1] = (int)(64u << 16);   // [47:32]=bar addr 0, [63:48]=tensor_dim0.lo=64
    g1[2] = (int)(64u << 16);   // [79:64]=tensor_dim0.hi=0, [95:80]=tensor_dim1.lo=64
    g1[3] = (int)(64u << 16);   // [111:96]=tensor_dim1.hi=0, [127:112]=tile_dim0=64
    g1[4] = 64;                 // [143:128]=tile_dim1=64, [159:144]=tile_dim2=0
    g1[5] = 64;                 // tensor_dim0_stride.lo = 64 (8B units)
    g1[6] = 0;
    g1[7] = 0;

    const i32x4 z4 = {0, 0, 0, 0};
    const i32x8 z8 = {0, 0, 0, 0, 0, 0, 0, 0};
    __builtin_amdgcn_tensor_load_to_lds(g0, g1, z4, z4, z8, 0);
}

// ---------------------------------------------------------------------------
// Kernel 1: row-normalize power_spectrum and convert support to bf16.
// One wave per 256-wide row.
// ---------------------------------------------------------------------------
__global__ __launch_bounds__(256) void gap_prep(const float* __restrict__ X,
                                                const float* __restrict__ S,
                                                unsigned short* __restrict__ Xb,
                                                unsigned short* __restrict__ Sb) {
    const int lane = threadIdx.x & 31;
    const int wave = threadIdx.x >> 5;
    const int row  = blockIdx.x * 8 + wave;

    const float* src;
    unsigned short* dst;
    bool norm;
    if (row < N_ATOMS) {
        src = X + (size_t)row * D_FEAT;
        dst = Xb + (size_t)row * D_FEAT;
        norm = true;
    } else {
        const int r = row - N_ATOMS;
        src = S + (size_t)r * D_FEAT;
        dst = Sb + (size_t)r * D_FEAT;
        norm = false;
    }

    float vals[8];
    float ss = 0.0f;
#pragma unroll
    for (int i = 0; i < 8; ++i) {
        vals[i] = src[lane + i * 32];
        ss += vals[i] * vals[i];
    }
#pragma unroll
    for (int off = 1; off < 32; off <<= 1)
        ss += __shfl_xor(ss, off, 32);

    const float scale = norm ? rsqrtf(ss) : 1.0f;
#pragma unroll
    for (int i = 0; i < 8; ++i)
        dst[lane + i * 32] = f2bf(vals[i] * scale);
}

// ---------------------------------------------------------------------------
// Kernel 2: wcol[j] = weights[col_segments[j]] ; zero the 512-float output.
// ---------------------------------------------------------------------------
__global__ __launch_bounds__(256) void gap_aux(const float* __restrict__ W,
                                               const int* __restrict__ colseg,
                                               float* __restrict__ wcol,
                                               float* __restrict__ out) {
    const int t = blockIdx.x * 256 + threadIdx.x;
    if (t < N_SUPPORT) wcol[t] = W[colseg[t]];
    if (t < N_STRUCT)  out[t]  = 0.0f;
}

// ---------------------------------------------------------------------------
// Kernel 3: fused GEMM + square + column-weight + row reduction.
// 256 blocks x 8 waves. Wave owns 16 rows (A resident in VGPRs for K=256).
// Support tiles staged into LDS by the Tensor Data Mover, double buffered:
// TDM copy of stage t+1 overlaps WMMA on stage t (TENSORcnt pipelining).
// Per stage a wave runs 4 independent 16x16 WMMA chains (64 columns).
// ---------------------------------------------------------------------------
__global__ __launch_bounds__(256) void gap_main(const unsigned short* __restrict__ Xb,
                                                const unsigned short* __restrict__ Sb,
                                                const float* __restrict__ wcol,
                                                const int* __restrict__ rowseg,
                                                float* __restrict__ out) {
    __shared__ __align__(16) unsigned char smem[2 * BUF_BYTES];

    const int lane  = threadIdx.x & 31;
    const int wave  = threadIdx.x >> 5;
    const int half  = lane >> 4;   // 0: lanes 0-15, 1: lanes 16-31
    const int l16   = lane & 15;
    const int mbase = blockIdx.x * 128 + wave * 16;

    // --- A fragments for all 8 K-steps, resident for the whole kernel ------
    const unsigned short* arow = Xb + (size_t)(mbase + l16) * D_FEAT;
    Frag a[8];
#pragma unroll
    for (int kk = 0; kk < 8; ++kk) {
        const unsigned short* p = arow + kk * 32 + half * 16;
        a[kk].u[0] = *(const u32x4*)(p);
        a[kk].u[1] = *(const u32x4*)(p + 8);
    }

    float acc[8];
#pragma unroll
    for (int r = 0; r < 8; ++r) acc[r] = 0.0f;

    const unsigned lds_base = (unsigned)(uintptr_t)(void*)&smem[0];

    // Prologue: kick off stage 0 into buffer 0 (one TDM issue per block).
    if (wave == 0) {
        tdm_load_tile(lds_base, Sb);
    }

#pragma unroll 1
    for (int t = 0; t < N_STAGES; ++t) {
        const int buf = t & 1;

        if (wave == 0) {
            if (t + 1 < N_STAGES) {
                tdm_load_tile(lds_base + (unsigned)((buf ^ 1) * BUF_BYTES),
                              Sb + (size_t)(t + 1) * STAGE_COLS * D_FEAT);
                __builtin_amdgcn_s_wait_tensorcnt((short)1); // stage t landed
            } else {
                __builtin_amdgcn_s_wait_tensorcnt((short)0);
            }
        }
        __syncthreads();   // publish staged tile to all waves

        const unsigned char* bufp = smem + buf * BUF_BYTES;

#pragma unroll
        for (int c = 0; c < 4; ++c) {
            // B 32x16 fragment: lane holds column (c*16 + l16);
            // K elements { half*16 + 0..15 } = 32 contiguous bytes in the row.
            const unsigned char* rowp =
                bufp + (size_t)(c * 16 + l16) * LDS_ROW + half * 32;

            v8f cc = {};
#pragma unroll
            for (int kk = 0; kk < 8; ++kk) {
                Frag b;
                const u32x4* p = (const u32x4*)(rowp + kk * 64);
                b.u[0] = p[0];
                b.u[1] = p[1];
                cc = __builtin_amdgcn_wmma_f32_16x16x32_bf16(
                        false, a[kk].v, false, b.v, (short)0, cc, false, false);
            }

            const float wn = wcol[t * STAGE_COLS + c * 16 + l16];
#pragma unroll
            for (int r = 0; r < 8; ++r)
                acc[r] += cc[r] * cc[r] * wn;
        }

        __syncthreads();   // all waves done with buf before it is re-filled
    }

    // --- reduce over the 16 columns held across each 16-lane half ----------
#pragma unroll
    for (int r = 0; r < 8; ++r) {
        float v = acc[r];
#pragma unroll
        for (int off = 1; off < 16; off <<= 1)
            v += __shfl_xor(v, off, 32);
        acc[r] = v;
    }

    // C f32 layout: VGPR r, lanes 0-15 -> row r; lanes 16-31 -> row r+8.
    if (l16 == 0) {
#pragma unroll
        for (int r = 0; r < 8; ++r) {
            const int row = mbase + r + half * 8;
            atomicAdd(&out[rowseg[row]], acc[r]);
        }
    }
}

// ---------------------------------------------------------------------------
extern "C" void kernel_launch(void* const* d_in, const int* in_sizes, int n_in,
                              void* d_out, int out_size, void* d_ws, size_t ws_size,
                              hipStream_t stream) {
    (void)in_sizes; (void)n_in; (void)out_size; (void)ws_size;

    const float* X      = (const float*)d_in[0];   // power_spectrum [32768,256]
    const float* S      = (const float*)d_in[1];   // support        [8192,256]
    const float* W      = (const float*)d_in[2];   // weights        [1,256]
    const int*   rowseg = (const int*)d_in[3];     // [32768]
    const int*   colseg = (const int*)d_in[4];     // [8192]
    float* out = (float*)d_out;                    // [512]

    unsigned short* Xb   = (unsigned short*)d_ws;                     // 16 MB
    unsigned short* Sb   = Xb + (size_t)N_ATOMS * D_FEAT;             //  4 MB
    float*          wcol = (float*)(Sb + (size_t)N_SUPPORT * D_FEAT); // 32 KB

    gap_prep<<<(N_ATOMS + N_SUPPORT) / 8, 256, 0, stream>>>(X, S, Xb, Sb);
    gap_aux<<<N_SUPPORT / 256, 256, 0, stream>>>(W, colseg, wcol, out);
    gap_main<<<N_ATOMS / 128, 256, 0, stream>>>(Xb, Sb, wcol, rowseg, out);
}
